// EnhancedCAREGNNLayer_53206054863626
// MI455X (gfx1250) — matrix operand order repaired and checked by reading, not compiled, not run on hardware
//
#include <hip/hip_runtime.h>
#include <math.h>

// ---------------- problem constants ----------------
constexpr int N_ = 100000, D_ = 128, O_ = 128, V_ = 3, E_ = 400000, C_ = 2;
constexpr int TILES = N_ / 16;   // 6250

typedef __attribute__((ext_vector_type(16))) __bf16 v16bf;
typedef __attribute__((ext_vector_type(8)))  float  v8f;

// ---------------- WMMA fragment helpers ----------------
// A fragment (16x32 bf16) from row-major f32, row = tile_base + m*ld.
// Lane L: m = L&15, g = L>>4. VGPR i<4: K = kk*32+g*8+2i..; VGPR 4+i: K = kk*32+16+g*8+2i..
__device__ __forceinline__ v16bf a_frag_f32(const float* row, int kk, int g) {
  const float4* h0 = (const float4*)(row + kk * 32 + g * 8);
  const float4* h1 = (const float4*)(row + kk * 32 + 16 + g * 8);
  float4 x0 = h0[0], x1 = h0[1], y0 = h1[0], y1 = h1[1];
  v16bf r;
  r[0]=(__bf16)x0.x; r[1]=(__bf16)x0.y; r[2]=(__bf16)x0.z; r[3]=(__bf16)x0.w;
  r[4]=(__bf16)x1.x; r[5]=(__bf16)x1.y; r[6]=(__bf16)x1.z; r[7]=(__bf16)x1.w;
  r[8]=(__bf16)y0.x; r[9]=(__bf16)y0.y; r[10]=(__bf16)y0.z; r[11]=(__bf16)y0.w;
  r[12]=(__bf16)y1.x; r[13]=(__bf16)y1.y; r[14]=(__bf16)y1.z; r[15]=(__bf16)y1.w;
  return r;
}

// Same but each element scaled by vp[K] (pe = view * view_pref along K axis)
__device__ __forceinline__ v16bf a_frag_f32_scaled(const float* row, const float* vp, int kk, int g) {
  const float4* h0 = (const float4*)(row + kk * 32 + g * 8);
  const float4* h1 = (const float4*)(row + kk * 32 + 16 + g * 8);
  const float4* s0 = (const float4*)(vp + kk * 32 + g * 8);
  const float4* s1 = (const float4*)(vp + kk * 32 + 16 + g * 8);
  float4 x0 = h0[0], x1 = h0[1], y0 = h1[0], y1 = h1[1];
  float4 a0 = s0[0], a1 = s0[1], b0 = s1[0], b1 = s1[1];
  v16bf r;
  r[0]=(__bf16)(x0.x*a0.x); r[1]=(__bf16)(x0.y*a0.y); r[2]=(__bf16)(x0.z*a0.z); r[3]=(__bf16)(x0.w*a0.w);
  r[4]=(__bf16)(x1.x*a1.x); r[5]=(__bf16)(x1.y*a1.y); r[6]=(__bf16)(x1.z*a1.z); r[7]=(__bf16)(x1.w*a1.w);
  r[8]=(__bf16)(y0.x*b0.x); r[9]=(__bf16)(y0.y*b0.y); r[10]=(__bf16)(y0.z*b0.z); r[11]=(__bf16)(y0.w*b0.w);
  r[12]=(__bf16)(y1.x*b1.x); r[13]=(__bf16)(y1.y*b1.y); r[14]=(__bf16)(y1.z*b1.z); r[15]=(__bf16)(y1.w*b1.w);
  return r;
}

// Full K=128 accumulation: 4 x wmma_f32_16x16x32_bf16.
// wrow points at wT row for this lane's output column (col*ld), kofs selects K window.
__device__ __forceinline__ v8f gemm_k128(const v16bf A[4], const __bf16* wrow, int g, int kofs, v8f acc) {
#pragma unroll
  for (int kk = 0; kk < 4; ++kk) {
    typedef __attribute__((ext_vector_type(8))) __bf16 v8bf_;
    v8bf_ b0 = *(const v8bf_*)(wrow + kofs + kk * 32 + g * 8);
    v8bf_ b1 = *(const v8bf_*)(wrow + kofs + kk * 32 + 16 + g * 8);
    v16bf b = __builtin_shufflevector(b0, b1, 0, 1, 2, 3, 4, 5, 6, 7, 8, 9, 10, 11, 12, 13, 14, 15);
    acc = __builtin_amdgcn_wmma_f32_16x16x32_bf16(false, A[kk], false, b, (short)0, acc, false, false);
  }
  return acc;
}

__device__ __forceinline__ float red16(float x) {
  x += __shfl_xor(x, 1, 32);
  x += __shfl_xor(x, 2, 32);
  x += __shfl_xor(x, 4, 32);
  x += __shfl_xor(x, 8, 32);
  return x;
}

__device__ __forceinline__ float sigmoidf(float x) { return 1.f / (1.f + __expf(-x)); }

__device__ __forceinline__ void atomAddF32(float* p, float v) {
  __hip_atomic_fetch_add(p, v, __ATOMIC_RELAXED, __HIP_MEMORY_SCOPE_AGENT);
}

// ---------------- K1: weight transpose + bf16 convert ----------------
__global__ __launch_bounds__(256) void k_prep_weights(
    const float* rel_w, const float* gate_w, const float* la_w1, const float* va_w1,
    const float* sl_w, const float* ft_w, const float* fus_w,
    __bf16* wrel, __bf16* wgate, __bf16* wla1, __bf16* wva1,
    __bf16* wsl, __bf16* wft, __bf16* wfus) {
  int i = blockIdx.x * 256 + threadIdx.x;
  if (i < V_ * 128 * 128) {                       // [v][o][d] <- rel_w[v][d][o]
    int v = i >> 14, o = (i >> 7) & 127, d = i & 127;
    wrel[i]  = (__bf16)rel_w[v * 16384 + d * 128 + o];
    wgate[i] = (__bf16)gate_w[v * 16384 + d * 128 + o];
  }
  if (i < C_ * 64 * 128) {                        // [c][k][d] <- la_w1[c][d][k]
    int c = i >> 13, k = (i >> 7) & 63, d = i & 127;
    wla1[i] = (__bf16)la_w1[c * 8192 + d * 64 + k];
  }
  if (i < 64 * 128) {                             // [k][o] <- va_w1[o][k]
    int k = i >> 7, d = i & 127;
    wva1[i] = (__bf16)va_w1[d * 64 + k];
  }
  if (i < 128 * 128) {                            // [o][d] <- w[d][o]
    int o = i >> 7, d = i & 127;
    wsl[i] = (__bf16)sl_w[d * 128 + o];
    wft[i] = (__bf16)ft_w[d * 128 + o];
  }
  if (i < 128 * 256) {                            // [o][i2] <- fus_w[i2][o], i2 in 0..255
    int o = i >> 8, d = i & 255;
    wfus[i] = (__bf16)fus_w[d * 128 + o];
  }
}

// ---------------- K2: per-node-tile dense projections (WMMA) ----------------
__global__ __launch_bounds__(32) void k_node_proj(
    const float* __restrict__ feat,
    const __bf16* __restrict__ wrel, const float* __restrict__ rel_b,
    const __bf16* __restrict__ wsl,  const float* __restrict__ sl_b,
    const __bf16* __restrict__ wft,  const float* __restrict__ ft_b,
    const __bf16* __restrict__ wla1, const float* __restrict__ la_b1,
    const float* __restrict__ la_w2, const float* __restrict__ la_b2,
    const float* __restrict__ pred_w, const float* __restrict__ pred_b,
    const float* __restrict__ att_bias,
    float* __restrict__ proj, float* __restrict__ self_out, float* __restrict__ trans,
    float* __restrict__ natt, float* __restrict__ cp_out) {
  const int tile = blockIdx.x;
  const int node0 = tile * 16;
  const int L = threadIdx.x;
  const int m = L & 15, g = L >> 4;

  __shared__ float s_sc[2][16];   // sigmoid scores per class per row
  __shared__ float s_cp[16][2];   // class probs per row

  // A fragments of the feature tile (K = 0..127)
  const float* frow = feat + (size_t)(node0 + m) * D_;
  v16bf A[4];
#pragma unroll
  for (int kk = 0; kk < 4; ++kk) A[kk] = a_frag_f32(frow, kk, g);

  // ---- 3 view projections: proj[v] = feat @ rel_w[v] + rel_b[v] ----
  for (int v = 0; v < V_; ++v) {
    const __bf16* w = wrel + v * 128 * 128;
#pragma unroll
    for (int t = 0; t < 8; ++t) {
      int col = t * 16 + m;
      v8f acc = {};
      acc = gemm_k128(A, w + col * 128, g, 0, acc);
      float b = rel_b[v * 128 + col];
      float* out = proj + ((size_t)v * N_ + node0) * 128 + col;
#pragma unroll
      for (int r = 0; r < 8; ++r) out[(size_t)(r + 8 * g) * 128] = acc[r] + b;
    }
  }

  // ---- self_out and transformed ----
#pragma unroll
  for (int t = 0; t < 8; ++t) {
    int col = t * 16 + m;
    v8f a1 = {}, a2 = {};
    a1 = gemm_k128(A, wsl + col * 128, g, 0, a1);
    a2 = gemm_k128(A, wft + col * 128, g, 0, a2);
    float b1 = sl_b[col], b2 = ft_b[col];
#pragma unroll
    for (int r = 0; r < 8; ++r) {
      size_t idx = (size_t)(node0 + r + 8 * g) * 128 + col;
      self_out[idx] = a1[r] + b1;
      trans[idx]    = a2[r] + b2;
    }
  }

  // ---- label-aware attention: h = relu(feat@la_w1[c]+b1); score=sigmoid(h.w2 + b2) ----
  for (int c = 0; c < C_; ++c) {
    float sp[8];
#pragma unroll
    for (int r = 0; r < 8; ++r) sp[r] = 0.f;
#pragma unroll
    for (int t = 0; t < 4; ++t) {
      int n = t * 16 + m;   // 0..63
      v8f acc = {};
      acc = gemm_k128(A, wla1 + (c * 64 + n) * 128, g, 0, acc);
      float b1 = la_b1[c * 64 + n], w2 = la_w2[c * 64 + n];
#pragma unroll
      for (int r = 0; r < 8; ++r) {
        float h = acc[r] + b1;
        h = h > 0.f ? h : 0.f;
        sp[r] += h * w2;
      }
    }
#pragma unroll
    for (int r = 0; r < 8; ++r) {
      float s = red16(sp[r]);
      if (m == r) s_sc[c][r + 8 * g] = sigmoidf(s + la_b2[c]);
    }
  }

  // ---- class probs: softmax(feat @ pred_w + pred_b) over C=2 ----
  {
    float p0 = 0.f, p1 = 0.f;
    const float* fr = feat + (size_t)(node0 + m) * D_ + g * 64;
    for (int d = 0; d < 64; ++d) {
      float f = fr[d];
      p0 += f * pred_w[(g * 64 + d) * 2 + 0];
      p1 += f * pred_w[(g * 64 + d) * 2 + 1];
    }
    p0 += __shfl_xor(p0, 16, 32);
    p1 += __shfl_xor(p1, 16, 32);
    p0 += pred_b[0];
    p1 += pred_b[1];
    float mx = fmaxf(p0, p1);
    float e0 = __expf(p0 - mx), e1 = __expf(p1 - mx);
    float inv = 1.f / (e0 + e1);
    e0 *= inv; e1 *= inv;
    if (g == 0) {
      cp_out[(size_t)(node0 + m) * 2 + 0] = e0;
      cp_out[(size_t)(node0 + m) * 2 + 1] = e1;
      s_cp[m][0] = e0;
      s_cp[m][1] = e1;
    }
  }
  __syncthreads();
  if (L < 16) {
    natt[node0 + L] = s_sc[0][L] * s_cp[L][0] + s_sc[1][L] * s_cp[L][1] + att_bias[0];
  }
}

// ---------------- K3: edge gather * weight -> atomic scatter-add ----------------
__global__ __launch_bounds__(256) void k_edge(
    const int* __restrict__ ei, const float* __restrict__ ew,
    const float* __restrict__ proj, float* __restrict__ agg) {
  int gid = blockIdx.x * 8 + (threadIdx.x >> 5);  // edge id over all views
  if (gid >= V_ * E_) return;
  int lane = threadIdx.x & 31;
  int v = gid / E_;
  int e = gid - v * E_;
  int src = ei[(size_t)v * 2 * E_ + e];
  int dst = ei[(size_t)v * 2 * E_ + E_ + e];
  float w = ew[(size_t)v * E_ + e];
  const float4* p = (const float4*)(proj + ((size_t)v * N_ + src) * 128) + lane;
  float4 val = *p;
  float* a = agg + ((size_t)v * N_ + dst) * 128 + lane * 4;
  atomAddF32(a + 0, val.x * w);
  atomAddF32(a + 1, val.y * w);
  atomAddF32(a + 2, val.z * w);
  atomAddF32(a + 3, val.w * w);
}

// ---------------- K4: gate -> views(LDS) -> view attention -> fusion -> layernorm ----------------
__global__ __launch_bounds__(32) void k_fuse(
    const float* __restrict__ agg,
    const __bf16* __restrict__ wgate, const float* __restrict__ gate_b,
    const float* __restrict__ view_pref,
    const __bf16* __restrict__ wva1, const float* __restrict__ va_b1,
    const float* __restrict__ va_w2, const float* __restrict__ va_b2,
    const float* __restrict__ self_out, const __bf16* __restrict__ wfus,
    const float* __restrict__ fus_b,
    const float* __restrict__ trans, const float* __restrict__ natt,
    const float* __restrict__ ln_g, const float* __restrict__ ln_b,
    float* __restrict__ out) {
  __shared__ float s_view[V_][16][128];
  __shared__ float s_w[16][128];

  const int tile = blockIdx.x;
  const int node0 = tile * 16;
  const int L = threadIdx.x;
  const int m = L & 15, g = L >> 4;

  // ---- 1) per view: view = sigmoid(agg@gate_w + gate_b) * agg ----
  for (int v = 0; v < V_; ++v) {
    const float* atile = agg + ((size_t)v * N_ + node0) * 128;
    const float* arow = atile + (size_t)m * 128;
    v16bf A[4];
#pragma unroll
    for (int kk = 0; kk < 4; ++kk) A[kk] = a_frag_f32(arow, kk, g);
    const __bf16* w = wgate + v * 128 * 128;
#pragma unroll
    for (int t = 0; t < 8; ++t) {
      int col = t * 16 + m;
      v8f acc = {};
      acc = gemm_k128(A, w + col * 128, g, 0, acc);
      float b = gate_b[v * 128 + col];
#pragma unroll
      for (int r = 0; r < 8; ++r) {
        int row = r + 8 * g;
        float av = atile[(size_t)row * 128 + col];
        s_view[v][row][col] = sigmoidf(acc[r] + b) * av;
      }
    }
  }
  __syncthreads();

  // ---- 2) view-level attention scores ----
  float vs[V_][8];
  for (int v = 0; v < V_; ++v) {
    v16bf A[4];
#pragma unroll
    for (int kk = 0; kk < 4; ++kk)
      A[kk] = a_frag_f32_scaled(&s_view[v][m][0], view_pref + v * 128, kk, g);
#pragma unroll
    for (int r = 0; r < 8; ++r) vs[v][r] = 0.f;
#pragma unroll
    for (int t = 0; t < 4; ++t) {
      int n = t * 16 + m;   // 0..63
      v8f acc = {};
      acc = gemm_k128(A, wva1 + n * 128, g, 0, acc);
      float b1 = va_b1[n], w2 = va_w2[n];
#pragma unroll
      for (int r = 0; r < 8; ++r) {
        float h = acc[r] + b1;
        h = h > 0.f ? h : 0.f;
        vs[v][r] += h * w2;
      }
    }
#pragma unroll
    for (int r = 0; r < 8; ++r) vs[v][r] = red16(vs[v][r]) + va_b2[0];
  }

  // softmax across views, per row
  float att[V_][8];
#pragma unroll
  for (int r = 0; r < 8; ++r) {
    float mx = fmaxf(vs[0][r], fmaxf(vs[1][r], vs[2][r]));
    float e0 = __expf(vs[0][r] - mx), e1 = __expf(vs[1][r] - mx), e2 = __expf(vs[2][r] - mx);
    float inv = 1.f / (e0 + e1 + e2);
    att[0][r] = e0 * inv; att[1][r] = e1 * inv; att[2][r] = e2 * inv;
  }

  // ---- 3) weighted = (sum_v att*view) * node_att ----
  float na[8];
#pragma unroll
  for (int r = 0; r < 8; ++r) na[r] = natt[node0 + r + 8 * g];
#pragma unroll
  for (int t = 0; t < 8; ++t) {
    int col = t * 16 + m;
#pragma unroll
    for (int r = 0; r < 8; ++r) {
      int row = r + 8 * g;
      float comb = att[0][r] * s_view[0][row][col] +
                   att[1][r] * s_view[1][row][col] +
                   att[2][r] * s_view[2][row][col];
      s_w[row][col] = comb * na[r];
    }
  }
  __syncthreads();

  // ---- 4) fusion: relu([self_out, weighted] @ fus_w + b) + trans ----
  v16bf Asf[4], Aw[4];
  const float* srow = self_out + (size_t)(node0 + m) * 128;
#pragma unroll
  for (int kk = 0; kk < 4; ++kk) {
    Asf[kk] = a_frag_f32(srow, kk, g);
    Aw[kk]  = a_frag_f32(&s_w[m][0], kk, g);
  }
  float o_[8][8];   // [col-tile][r]
#pragma unroll
  for (int t = 0; t < 8; ++t) {
    int col = t * 16 + m;
    v8f acc = {};
    acc = gemm_k128(Asf, wfus + col * 256, g, 0, acc);     // K 0..127 (self part)
    acc = gemm_k128(Aw,  wfus + col * 256, g, 128, acc);   // K 128..255 (weighted part)
    float b = fus_b[col];
#pragma unroll
    for (int r = 0; r < 8; ++r) {
      float f = acc[r] + b;
      f = f > 0.f ? f : 0.f;
      o_[t][r] = f + trans[(size_t)(node0 + r + 8 * g) * 128 + col];
    }
  }

  // ---- 5) layernorm over 128 cols per row ----
#pragma unroll
  for (int r = 0; r < 8; ++r) {
    float s = 0.f;
#pragma unroll
    for (int t = 0; t < 8; ++t) s += o_[t][r];
    float mu = red16(s) * (1.f / 128.f);
    float q = 0.f;
#pragma unroll
    for (int t = 0; t < 8; ++t) { float d = o_[t][r] - mu; q += d * d; }
    float var = red16(q) * (1.f / 128.f);
    float inv = rsqrtf(var + 1e-5f);
#pragma unroll
    for (int t = 0; t < 8; ++t) {
      int col = t * 16 + m;
      out[(size_t)(node0 + r + 8 * g) * 128 + col] =
          (o_[t][r] - mu) * inv * ln_g[col] + ln_b[col];
    }
  }
}

// ---------------- launch ----------------
extern "C" void kernel_launch(void* const* d_in, const int* in_sizes, int n_in,
                              void* d_out, int out_size, void* d_ws, size_t ws_size,
                              hipStream_t stream) {
  (void)in_sizes; (void)n_in; (void)out_size; (void)ws_size;

  const float* features   = (const float*)d_in[0];
  const int*   edge_index = (const int*)d_in[1];
  const float* edge_weight= (const float*)d_in[2];
  const float* rel_w  = (const float*)d_in[3];
  const float* rel_b  = (const float*)d_in[4];
  const float* gate_w = (const float*)d_in[5];
  const float* gate_b = (const float*)d_in[6];
  const float* la_w1  = (const float*)d_in[7];
  const float* la_b1  = (const float*)d_in[8];
  const float* la_w2  = (const float*)d_in[9];
  const float* la_b2  = (const float*)d_in[10];
  const float* pred_w = (const float*)d_in[11];
  const float* pred_b = (const float*)d_in[12];
  const float* att_bias = (const float*)d_in[13];
  const float* view_pref= (const float*)d_in[14];
  const float* va_w1  = (const float*)d_in[15];
  const float* va_b1  = (const float*)d_in[16];
  const float* va_w2  = (const float*)d_in[17];
  const float* va_b2  = (const float*)d_in[18];
  const float* ft_w   = (const float*)d_in[19];
  const float* ft_b   = (const float*)d_in[20];
  const float* sl_w   = (const float*)d_in[21];
  const float* sl_b   = (const float*)d_in[22];
  const float* fus_w  = (const float*)d_in[23];
  const float* fus_b  = (const float*)d_in[24];
  const float* ln_g   = (const float*)d_in[25];
  const float* ln_beta= (const float*)d_in[26];

  // workspace layout (bytes)
  char* ws = (char*)d_ws;
  size_t off = 0;
  __bf16* wrel  = (__bf16*)(ws + off); off += (size_t)V_ * 128 * 128 * 2;  // 98304
  __bf16* wgate = (__bf16*)(ws + off); off += (size_t)V_ * 128 * 128 * 2;
  __bf16* wla1  = (__bf16*)(ws + off); off += (size_t)C_ * 64 * 128 * 2;
  __bf16* wva1  = (__bf16*)(ws + off); off += (size_t)64 * 128 * 2;
  __bf16* wsl   = (__bf16*)(ws + off); off += (size_t)128 * 128 * 2;
  __bf16* wft   = (__bf16*)(ws + off); off += (size_t)128 * 128 * 2;
  __bf16* wfus  = (__bf16*)(ws + off); off += (size_t)128 * 256 * 2;
  float* proj   = (float*)(ws + off);  off += (size_t)V_ * N_ * 128 * 4;   // 153.6 MB
  float* aggb   = (float*)(ws + off);  size_t agg_bytes = (size_t)V_ * N_ * 128 * 4;
  off += agg_bytes;
  float* selfb  = (float*)(ws + off);  off += (size_t)N_ * 128 * 4;
  float* transb = (float*)(ws + off);  off += (size_t)N_ * 128 * 4;
  float* nattb  = (float*)(ws + off);  off += (size_t)N_ * 4;

  float* out_main = (float*)d_out;                 // [N,128]
  float* out_cp   = out_main + (size_t)N_ * 128;   // [N,2]

  hipMemsetAsync(aggb, 0, agg_bytes, stream);

  k_prep_weights<<<192, 256, 0, stream>>>(rel_w, gate_w, la_w1, va_w1, sl_w, ft_w, fus_w,
                                          wrel, wgate, wla1, wva1, wsl, wft, wfus);

  k_node_proj<<<TILES, 32, 0, stream>>>(features,
                                        wrel, rel_b, wsl, sl_b, wft, ft_b,
                                        wla1, la_b1, la_w2, la_b2,
                                        pred_w, pred_b, att_bias,
                                        proj, selfb, transb, nattb, out_cp);

  k_edge<<<(V_ * E_) / 8, 256, 0, stream>>>(edge_index, edge_weight, proj, aggb);

  k_fuse<<<TILES, 32, 0, stream>>>(aggb, wgate, gate_b,
                                   view_pref, wva1, va_b1, va_w2, va_b2,
                                   selfb, wfus, fus_b,
                                   transb, nattb, ln_g, ln_beta,
                                   out_main);
}
